// SimpleGCN_40458591928693
// MI455X (gfx1250) — matrix operand order repaired and checked by reading, not compiled, not run on hardware
//
#include <hip/hip_runtime.h>
#include <hip/hip_fp16.h>

typedef __attribute__((ext_vector_type(16))) _Float16 v16h;
typedef __attribute__((ext_vector_type(8)))  _Float16 v8h;
typedef __attribute__((ext_vector_type(4)))  _Float16 v4h;
typedef __attribute__((ext_vector_type(8)))  float    v8f;

#define HDIM 64

// ---------------- degree / norm ----------------
__global__ void k_init_deg(float* deg, int n) {
    int i = blockIdx.x * blockDim.x + threadIdx.x;
    if (i < n) deg[i] = 1.0f;  // self-loop contribution
}

__global__ void k_count_deg(const int* __restrict__ dst, float* deg, int e) {
    int i = blockIdx.x * blockDim.x + threadIdx.x;
    if (i < e) atomicAdd(&deg[dst[i]], 1.0f);
}

// dis = rsqrt(deg) in-place; s init with self-loop term x[i]*dis^2
__global__ void k_dis_s(const float* __restrict__ x, float* degdis, float* s, int n) {
    int i = blockIdx.x * blockDim.x + threadIdx.x;
    if (i < n) {
        float d = rsqrtf(degdis[i]);   // deg >= 1 always (self loops)
        degdis[i] = d;
        s[i] = x[i] * d * d;
    }
}

// layer-1 scalar aggregation: s[dst] += x[src]*dis[src]*dis[dst]
__global__ void k_agg1(const int* __restrict__ src, const int* __restrict__ dst,
                       const float* __restrict__ x, const float* __restrict__ dis,
                       float* s, int e) {
    int i = blockIdx.x * blockDim.x + threadIdx.x;
    if (i < e) {
        int sr = src[i], d = dst[i];
        atomicAdd(&s[d], x[sr] * dis[sr] * dis[d]);
    }
}

// h1[i][j] = relu(s[i]*W1[j] + b1[j])  (stored fp16 for WMMA)
__global__ void k_h1(const float* __restrict__ s, const float* __restrict__ W1,
                     const float* __restrict__ b1, _Float16* __restrict__ h1, int n64) {
    int idx = blockIdx.x * blockDim.x + threadIdx.x;
    if (idx < n64) {
        int i = idx >> 6, j = idx & 63;
        float v = s[i] * W1[j] + b1[j];
        h1[idx] = (_Float16)fmaxf(v, 0.0f);
    }
}

// pack W2 (f32, 64x64 row-major) into WMMA B fragments.
// Bpack layout: [nt(4)][kt(2)][lane(32)][16 f16]  contiguous per lane.
// B 32x16 16-bit layout: lanes 0-15 -> N=lane, K = e (e<8), 8+e (e>=8); lanes 16-31 -> N=lane-16, K = 8+e (e<8), 16+e (e>=8).
__global__ void k_packB(const float* __restrict__ W2, _Float16* __restrict__ Bpack) {
    int t = threadIdx.x;            // 256 threads = 8 frags * 32 lanes
    int lane = t & 31;
    int frag = t >> 5;              // frag = nt*2 + kt
    int kt = frag & 1;
    int nt = frag >> 1;
    int ncol = nt * 16 + (lane & 15);
    int kbase = kt * 32 + ((lane >= 16) ? 8 : 0);
#pragma unroll
    for (int e = 0; e < 16; e++)
        Bpack[t * 16 + e] = (_Float16)W2[(kbase + e + ((e >= 8) ? 8 : 0)) * HDIM + ncol];
}

// WMMA GEMM over FULL 16-row tiles: t = h1 @ W2
// One wave computes a 16x64 tile with 8x v_wmma_f32_16x16x32_f16, stages the
// result through LDS, and stores it back with coalesced b128 stores.
__global__ void k_gemm(const _Float16* __restrict__ h1, const _Float16* __restrict__ Bpack,
                       _Float16* __restrict__ tout, int fullTiles) {
    __shared__ _Float16 ct[8][16][72];   // per-wave 16x64 tile, padded row (144B, 16B aligned)

    int wave = (blockIdx.x * blockDim.x + threadIdx.x) >> 5;
    int lane = threadIdx.x & 31;
    int w = threadIdx.x >> 5;
    if (wave >= fullTiles) return;       // wave-uniform: EXEC all-1s for WMMA
    int rowBase = wave << 4;
    int mrow = rowBase + (lane & 15);
    int kofs = (lane >= 16) ? 8 : 0;

    // A fragments (16-bit 16x32 layout): lane holds K chunks [kofs..kofs+7] and [16+kofs..+7]
    const _Float16* arow = h1 + (size_t)mrow * HDIM;
    v8h c00 = *(const v8h*)(arow + 0  + kofs);
    v8h c01 = *(const v8h*)(arow + 16 + kofs);
    v8h c10 = *(const v8h*)(arow + 32 + kofs);
    v8h c11 = *(const v8h*)(arow + 48 + kofs);
    v16h a0, a1;
#pragma unroll
    for (int e = 0; e < 8; e++) {
        a0[e] = c00[e]; a0[8 + e] = c01[e];
        a1[e] = c10[e]; a1[8 + e] = c11[e];
    }

    const v16h* B = (const v16h*)Bpack;
    int colw = lane & 15;
    int mloc = (lane >= 16) ? 8 : 0;
#pragma unroll
    for (int nt = 0; nt < 4; nt++) {
        v8f c = {};
        c = __builtin_amdgcn_wmma_f32_16x16x32_f16(false, a0, false, B[(nt * 2 + 0) * 32 + lane],
                                                   (short)0, c, false, false);
        c = __builtin_amdgcn_wmma_f32_16x16x32_f16(false, a1, false, B[(nt * 2 + 1) * 32 + lane],
                                                   (short)0, c, false, false);
#pragma unroll
        for (int v = 0; v < 8; v++)      // C/D layout: VGPR v -> M = mloc+v, N = nt*16+colw
            ct[w][mloc + v][nt * 16 + colw] = (_Float16)c[v];
    }

    __builtin_amdgcn_wave_barrier();     // wave-internal LDS reuse; DS ops are in-order per wave

    // each lane copies half a row (32 f16 = 64B) out, coalesced
    int rloc = lane & 15;
    int coff = (lane >> 4) * 32;
    const _Float16* lsrc = &ct[w][rloc][coff];
    _Float16* gdst = tout + (size_t)(rowBase + rloc) * HDIM + coff;
    v8h s0 = *(const v8h*)(lsrc + 0);
    v8h s1 = *(const v8h*)(lsrc + 8);
    v8h s2 = *(const v8h*)(lsrc + 16);
    v8h s3 = *(const v8h*)(lsrc + 24);
    *(v8h*)(gdst + 0)  = s0;
    *(v8h*)(gdst + 8)  = s1;
    *(v8h*)(gdst + 16) = s2;
    *(v8h*)(gdst + 24) = s3;
}

// scalar fallback for remainder rows (host launches only if N % 16 != 0)
__global__ void k_gemm_rem(const _Float16* __restrict__ h1, const float* __restrict__ W2,
                           _Float16* __restrict__ tout, int startRow, int n) {
    int idx = blockIdx.x * blockDim.x + threadIdx.x;
    int r = startRow + (idx >> 6);
    int c = idx & 63;
    if (r < n) {
        float acc = 0.0f;
#pragma unroll 8
        for (int k = 0; k < HDIM; k++)
            acc += (float)h1[(size_t)r * HDIM + k] * W2[k * HDIM + c];
        tout[(size_t)r * HDIM + c] = (_Float16)acc;
    }
}

// h2acc init with self-loop: h2acc[i][j] = t[i][j] * dis[i]^2
__global__ void k_init_h2(const _Float16* __restrict__ t, const float* __restrict__ dis,
                          float* __restrict__ h2acc, int n64) {
    int idx = blockIdx.x * blockDim.x + threadIdx.x;
    if (idx < n64) {
        float d = dis[idx >> 6];
        h2acc[idx] = (float)t[idx] * d * d;
    }
}

// layer-2 vector aggregation: each thread handles one edge x 4 columns
__global__ void k_agg2(const int* __restrict__ src, const int* __restrict__ dst,
                       const float* __restrict__ dis, const _Float16* __restrict__ t,
                       float* h2acc, int e16) {
    int idx = blockIdx.x * blockDim.x + threadIdx.x;
    if (idx < e16) {
        int ed = idx >> 4, q = (idx & 15) << 2;
        int sr = src[ed], d = dst[ed];
        float nrm = dis[sr] * dis[d];
        v4h vv = *(const v4h*)(t + (size_t)sr * HDIM + q);
        float* hd = h2acc + (size_t)d * HDIM + q;
#pragma unroll
        for (int c = 0; c < 4; c++)
            atomicAdd(&hd[c], (float)vv[c] * nrm);
    }
}

__global__ void k_zero_g(float* gsum, float* gcnt, int g) {
    int i = blockIdx.x * blockDim.x + threadIdx.x;
    if (i < g) { gsum[i] = 0.0f; gcnt[i] = 0.0f; }
}

// fused relu(h2acc+b2) . Wfc  -> per-graph sum & count (mean-pool and FC commute)
__global__ void k_pool(const float* __restrict__ h2acc, const float* __restrict__ b2,
                       const float* __restrict__ Wfc, const int* __restrict__ batch,
                       float* gsum, float* gcnt, int n) {
    int i = blockIdx.x * blockDim.x + threadIdx.x;
    if (i < n) {
        const float* row = h2acc + (size_t)i * HDIM;
        float acc = 0.0f;
#pragma unroll 8
        for (int j = 0; j < HDIM; j++) {
            float v = fmaxf(row[j] + b2[j], 0.0f);
            acc += v * Wfc[j];
        }
        int g = batch[i];
        atomicAdd(&gsum[g], acc);
        atomicAdd(&gcnt[g], 1.0f);
    }
}

__global__ void k_out(const float* __restrict__ gsum, const float* __restrict__ gcnt,
                      const float* __restrict__ bfc, float* out, int g) {
    int i = blockIdx.x * blockDim.x + threadIdx.x;
    if (i < g) out[i] = gsum[i] / fmaxf(gcnt[i], 1.0f) + bfc[0];
}

static inline size_t align256(size_t x) { return (x + 255) & ~(size_t)255; }

extern "C" void kernel_launch(void* const* d_in, const int* in_sizes, int n_in,
                              void* d_out, int out_size, void* d_ws, size_t ws_size,
                              hipStream_t stream) {
    const float* x    = (const float*)d_in[0];
    const int*   ei   = (const int*)d_in[1];
    const int*   bat  = (const int*)d_in[2];
    const float* W1   = (const float*)d_in[3];
    const float* b1   = (const float*)d_in[4];
    const float* W2   = (const float*)d_in[5];
    const float* b2   = (const float*)d_in[6];
    const float* Wfc  = (const float*)d_in[7];
    const float* bfc  = (const float*)d_in[8];
    float* out = (float*)d_out;

    const int N = in_sizes[0];
    const int E = in_sizes[1] / 2;
    const int G = out_size;
    const int* esrc = ei;
    const int* edst = ei + E;

    // workspace layout
    char* w = (char*)d_ws;
    size_t off = 0;
    float* deg = (float*)(w + off);      off = align256(off + (size_t)N * 4);         // deg -> dis
    float* s   = (float*)(w + off);      off = align256(off + (size_t)N * 4);
    _Float16* h1 = (_Float16*)(w + off); off = align256(off + (size_t)N * HDIM * 2);
    _Float16* Bp = (_Float16*)(w + off); off = align256(off + (size_t)4 * 2 * 32 * 16 * 2);
    _Float16* t  = (_Float16*)(w + off); off = align256(off + (size_t)N * HDIM * 2);
    float* h2acc = (float*)(w + off);    off = align256(off + (size_t)N * HDIM * 4);
    float* gsum  = (float*)(w + off);    off = align256(off + (size_t)G * 4);
    float* gcnt  = (float*)(w + off);    off = align256(off + (size_t)G * 4);
    (void)ws_size; (void)n_in;

    const int B = 256;
    const int n64 = N * HDIM;
    const int e16 = E * 16;
    const int fullTiles = N >> 4;        // whole 16-row tiles for the WMMA path
    const int remRows = N & 15;

    k_init_deg<<<(N + B - 1) / B, B, 0, stream>>>(deg, N);
    k_count_deg<<<(E + B - 1) / B, B, 0, stream>>>(edst, deg, E);
    k_dis_s<<<(N + B - 1) / B, B, 0, stream>>>(x, deg, s, N);
    k_agg1<<<(E + B - 1) / B, B, 0, stream>>>(esrc, edst, x, deg, s, E);
    k_h1<<<(n64 + B - 1) / B, B, 0, stream>>>(s, W1, b1, h1, n64);
    k_packB<<<1, 256, 0, stream>>>(W2, Bp);
    if (fullTiles > 0)
        k_gemm<<<(fullTiles * 32 + B - 1) / B, B, 0, stream>>>(h1, Bp, t, fullTiles);
    if (remRows > 0)
        k_gemm_rem<<<(remRows * HDIM + B - 1) / B, B, 0, stream>>>(h1, W2, t, fullTiles * 16, N);
    k_init_h2<<<(n64 + B - 1) / B, B, 0, stream>>>(t, deg, h2acc, n64);
    k_agg2<<<(e16 + B - 1) / B, B, 0, stream>>>(esrc, edst, deg, t, h2acc, e16);
    k_zero_g<<<(G + B - 1) / B, B, 0, stream>>>(gsum, gcnt, G);
    k_pool<<<(N + B - 1) / B, B, 0, stream>>>(h2acc, b2, Wfc, bat, gsum, gcnt, N);
    k_out<<<(G + B - 1) / B, B, 0, stream>>>(gsum, gcnt, bfc, out, G);
}
